// DynamicConvolution_66726611911196
// MI455X (gfx1250) — matrix operand order repaired and checked by reading, not compiled, not run on hardware
//
#include <hip/hip_runtime.h>

typedef unsigned short ushort_t;
typedef __attribute__((ext_vector_type(16))) __bf16 v16bf;
typedef __attribute__((ext_vector_type(8)))  float  v8f;
typedef __attribute__((ext_vector_type(4)))  int    v4i;

#define B_     32
#define CIN    128
#define COUT   128
#define HH     64
#define WW     64
#define KK     8
#define HID    512
#define NTAP   9

// ---- fp32 -> bf16 (round to nearest even) as raw bits ----
__device__ __forceinline__ ushort_t f2bf(float f) {
    unsigned u = __float_as_uint(f);
    unsigned r = u + 0x7FFFu + ((u >> 16) & 1u);
    return (ushort_t)(r >> 16);
}

union V16 { v4i i2[2]; v16bf bf; };

// ============ Kernel 1: x -> channels-last bf16 (+ fused per-row pool partials) ============
// One block per (b, y). Reads x row-major coalesced, transposes via padded LDS,
// writes channels-last bf16. Then re-reads the (cache-hot) 32KB slab to produce
// exact fp32 per-(ci) row sums -> part[b][y][ci]. Saves a full 67MB pooling pass.
__global__ __launch_bounds__(256) void xt_pool_kernel(const float* __restrict__ x,
                                                      ushort_t* __restrict__ xt,
                                                      float* __restrict__ part) {
    __shared__ ushort_t tile[WW * 130];                 // pad stride 130 to dodge bank conflicts
    const int b = blockIdx.x >> 6;
    const int y = blockIdx.x & 63;
    // read coalesced along w for each ci
    for (int it = 0; it < (CIN * WW) / 256; ++it) {
        int idx = it * 256 + threadIdx.x;
        int ci = idx >> 6, w = idx & 63;
        tile[w * 130 + ci] = f2bf(x[(((size_t)b * CIN + ci) * HH + y) * WW + w]);
    }
    __syncthreads();
    // write coalesced along ci for each w
    for (int it = 0; it < (CIN * WW) / 256; ++it) {
        int idx = it * 256 + threadIdx.x;
        int w = idx >> 7, ci = idx & 127;
        xt[(((size_t)b * HH + y) * WW + w) * CIN + ci] = tile[w * 130 + ci];
    }
    // fp32 row sums (reads hit WGP cache; no extra HBM traffic)
    if (threadIdx.x < CIN) {
        const float* p = x + (((size_t)b * CIN + threadIdx.x) * HH + y) * WW;
        float s = 0.f;
        for (int w = 0; w < WW; ++w) s += p[w];
        part[((size_t)b * HH + y) * CIN + threadIdx.x] = s;
    }
}

// ============ Kernel 2: finish pooling: reduce row partials over y ============
__global__ __launch_bounds__(256) void pool_finish_kernel(const float* __restrict__ part,
                                                          float* __restrict__ pooled) {
    const int g = blockIdx.x * 256 + threadIdx.x;       // b*CIN + ci, total 4096
    const int b = g >> 7, ci = g & 127;
    float s = 0.f;
    for (int y = 0; y < HH; ++y) s += part[((size_t)b * HH + y) * CIN + ci];
    pooled[g] = s * (1.0f / (HH * WW));
}

// ============ Kernel 3: MLP + prompt scores + softmax + agg bias, one block per b ============
__global__ __launch_bounds__(512) void attn_kernel(const float* __restrict__ pooled,
                                                   const float* __restrict__ prompt,
                                                   const float* __restrict__ w1, const float* __restrict__ b1,
                                                   const float* __restrict__ w2, const float* __restrict__ b2,
                                                   const float* __restrict__ kbias,
                                                   float* __restrict__ alphas,
                                                   float* __restrict__ aggb) {
    __shared__ float sp[CIN];
    __shared__ float sh[HID];
    __shared__ float ss[HID];
    __shared__ float sc[KK];
    const int b = blockIdx.x, t = threadIdx.x;
    if (t < CIN) sp[t] = pooled[b * CIN + t];
    __syncthreads();
    {   // h = relu(pooled @ w1.T + b1)
        float a = b1[t];
        for (int i = 0; i < CIN; ++i) a += sp[i] * w1[t * CIN + i];
        sh[t] = a > 0.f ? a : 0.f;
    }
    __syncthreads();
    {   // s = h @ w2.T + b2
        float a = b2[t];
        for (int i = 0; i < HID; ++i) a += sh[i] * w2[t * HID + i];
        ss[t] = a;
    }
    __syncthreads();
    if (t < KK) {                               // scores
        float a = 0.f;
        for (int i = 0; i < HID; ++i) a += ss[i] * prompt[t * HID + i];
        sc[t] = a;
    }
    __syncthreads();
    if (t == 0) {                               // softmax (TEMPERATURE = 1)
        float m = sc[0];
        for (int k = 1; k < KK; ++k) m = fmaxf(m, sc[k]);
        float sum = 0.f;
        for (int k = 0; k < KK; ++k) { sc[k] = __expf(sc[k] - m); sum += sc[k]; }
        float inv = 1.0f / sum;
        for (int k = 0; k < KK; ++k) sc[k] *= inv;
    }
    __syncthreads();
    if (t < KK) alphas[b * KK + t] = sc[t];
    if (t < COUT) {
        float a = 0.f;
        for (int k = 0; k < KK; ++k) a += sc[k] * kbias[k * COUT + t];
        aggb[b * COUT + t] = a;
    }
}

// ============ Kernel 4: aggregate weights -> bf16, layout [b][tap][o][ci] ============
__global__ __launch_bounds__(256) void aggw_kernel(const float* __restrict__ alphas,
                                                   const float* __restrict__ kw,
                                                   ushort_t* __restrict__ wt) {
    const int idx = blockIdx.x * 256 + threadIdx.x;     // total = B*9*COUT*CIN
    const int ci  = idx & (CIN - 1);
    const int o   = (idx >> 7) & (COUT - 1);
    const int tap = (idx >> 14) % NTAP;
    const int b   = idx / (CIN * COUT * NTAP);
    const int kh = tap / 3, kwv = tap % 3;
    float a = 0.f;
    #pragma unroll
    for (int k = 0; k < KK; ++k) {
        a += alphas[b * KK + k] *
             kw[((((size_t)k * COUT + o) * CIN + ci) * 3 + kh) * 3 + kwv];
    }
    wt[idx] = f2bf(a);
}

// ============ Kernel 5: per-sample 3x3 conv, implicit GEMM, 2x2 register blocking ============
// Each wave: 32 out-channels x 32 pixels via 4 fp32 accumulators. Per tap/ci-chunk,
// 2 A + 2 B operand load-pairs feed 4 v_wmma_f32_16x16x32_bf16 (2 b128 loads per WMMA).
__global__ __launch_bounds__(256) void conv_wmma_kernel(const ushort_t* __restrict__ xt,
                                                        const ushort_t* __restrict__ wt,
                                                        const float* __restrict__ aggb,
                                                        float* __restrict__ out) {
    const int lane = threadIdx.x & 31;
    const int wave = threadIdx.x >> 5;
    const int task = blockIdx.x * 8 + wave;    // B * 4 opairs * 128 ppairs = 16384
    const int b     = task >> 9;
    const int rem   = task & 511;
    const int opair = rem >> 7;                // 0..3 -> out channels [opair*32, +32)
    const int ppair = rem & 127;               // 0..127 -> 32 pixels of one row
    const int y  = ppair >> 1;
    const int x0 = (ppair & 1) << 5;
    const int n  = lane & 15;
    const int hi = lane >> 4;
    const int px0 = x0 + n;
    const int px1 = px0 + 16;

    const int o_lane0 = opair * 32 + n;        // A0: M = lane%16 out-channel row
    const int khalfA = hi ? 8 : 0;             // A K sub-offset per ISA layout
    const int kbaseB = hi ? 8 : 0;             // B K sub-offset per ISA layout

    const ushort_t* wbase0 = wt + (((size_t)b * NTAP) * COUT + o_lane0) * CIN;
    const ushort_t* wbase1 = wbase0 + (size_t)16 * CIN;      // second o-tile

    v8f acc00 = {}, acc01 = {}, acc10 = {}, acc11 = {};

    for (int tap = 0; tap < NTAP; ++tap) {
        const int dy = tap / 3 - 1, dx = tap % 3 - 1;
        const int yy = y + dy;
        const int xx0 = px0 + dx, xx1 = px1 + dx;
        const bool vy = (yy >= 0) & (yy < HH);
        const bool valid0 = vy & (xx0 >= 0) & (xx0 < WW);
        const bool valid1 = vy & (xx1 >= 0) & (xx1 < WW);
        const ushort_t* xrow0 =
            xt + (((size_t)b * HH + (valid0 ? yy : 0)) * WW + (valid0 ? xx0 : 0)) * CIN;
        const ushort_t* xrow1 =
            xt + (((size_t)b * HH + (valid1 ? yy : 0)) * WW + (valid1 ? xx1 : 0)) * CIN;
        const ushort_t* wrow0 = wbase0 + (size_t)tap * COUT * CIN;
        const ushort_t* wrow1 = wbase1 + (size_t)tap * COUT * CIN;
        #pragma unroll
        for (int c0 = 0; c0 < CIN; c0 += 32) {
            V16 ua0, ua1;   // weights: two contiguous 16B runs of K per operand
            ua0.i2[0] = *(const v4i*)(wrow0 + c0 + khalfA);
            ua0.i2[1] = *(const v4i*)(wrow0 + c0 + 16 + khalfA);
            ua1.i2[0] = *(const v4i*)(wrow1 + c0 + khalfA);
            ua1.i2[1] = *(const v4i*)(wrow1 + c0 + 16 + khalfA);
            V16 ub0, ub1;   // pixels: two contiguous 16B runs of K per operand
            if (valid0) {
                ub0.i2[0] = *(const v4i*)(xrow0 + c0 + kbaseB);
                ub0.i2[1] = *(const v4i*)(xrow0 + c0 + kbaseB + 16);
            } else {
                v4i z = {}; ub0.i2[0] = z; ub0.i2[1] = z;
            }
            if (valid1) {
                ub1.i2[0] = *(const v4i*)(xrow1 + c0 + kbaseB);
                ub1.i2[1] = *(const v4i*)(xrow1 + c0 + kbaseB + 16);
            } else {
                v4i z = {}; ub1.i2[0] = z; ub1.i2[1] = z;
            }
            acc00 = __builtin_amdgcn_wmma_f32_16x16x32_bf16(
                false, ua0.bf, false, ub0.bf, (short)0, acc00, false, false);
            acc01 = __builtin_amdgcn_wmma_f32_16x16x32_bf16(
                false, ua0.bf, false, ub1.bf, (short)0, acc01, false, false);
            acc10 = __builtin_amdgcn_wmma_f32_16x16x32_bf16(
                false, ua1.bf, false, ub0.bf, (short)0, acc10, false, false);
            acc11 = __builtin_amdgcn_wmma_f32_16x16x32_bf16(
                false, ua1.bf, false, ub1.bf, (short)0, acc11, false, false);
        }
    }

    // D layout: lane half hi -> M = v + 8*hi ; N = lane%16 = pixel
    const int obase0 = opair * 32 + hi * 8;
    const int obase1 = obase0 + 16;
    #pragma unroll
    for (int v = 0; v < 8; ++v) {
        const int oA = obase0 + v;
        const int oB = obase1 + v;
        const float bA = aggb[b * COUT + oA];
        const float bB = aggb[b * COUT + oB];
        out[(((size_t)b * COUT + oA) * HH + y) * WW + px0] = acc00[v] + bA;
        out[(((size_t)b * COUT + oA) * HH + y) * WW + px1] = acc01[v] + bA;
        out[(((size_t)b * COUT + oB) * HH + y) * WW + px0] = acc10[v] + bB;
        out[(((size_t)b * COUT + oB) * HH + y) * WW + px1] = acc11[v] + bB;
    }
}

// ============ launcher ============
extern "C" void kernel_launch(void* const* d_in, const int* in_sizes, int n_in,
                              void* d_out, int out_size, void* d_ws, size_t ws_size,
                              hipStream_t stream) {
    const float* x      = (const float*)d_in[0];
    const float* prompt = (const float*)d_in[1];
    const float* w1     = (const float*)d_in[2];
    const float* b1     = (const float*)d_in[3];
    const float* w2     = (const float*)d_in[4];
    const float* b2     = (const float*)d_in[5];
    const float* kw     = (const float*)d_in[6];
    const float* kb     = (const float*)d_in[7];
    float* out          = (float*)d_out;

    char* ws = (char*)d_ws;
    // workspace layout (bytes)
    float*    pooled = (float*)(ws);                              // 32*128*4        = 16384
    float*    alphas = (float*)(ws + 16384);                      // 32*8*4          = 1024
    float*    aggb   = (float*)(ws + 17408);                      // 32*128*4        = 16384
    float*    part   = (float*)(ws + 36864);                      // 32*64*128*4     = 1048576
    ushort_t* xt     = (ushort_t*)(ws + 36864 + 1048576);         // 32*64*64*128*2  = 33554432
    ushort_t* wt     = (ushort_t*)(ws + 36864 + 1048576 + 33554432); // 32*9*128*128*2 = 9437184

    xt_pool_kernel<<<B_ * HH, 256, 0, stream>>>(x, xt, part);
    pool_finish_kernel<<<(B_ * CIN) / 256, 256, 0, stream>>>(part, pooled);
    attn_kernel<<<B_, 512, 0, stream>>>(pooled, prompt, w1, b1, w2, b2, kb, alphas, aggb);
    aggw_kernel<<<(B_ * NTAP * COUT * CIN) / 256, 256, 0, stream>>>(alphas, kw, wt);
    conv_wmma_kernel<<<(B_ * 4 * 128) / 8, 256, 0, stream>>>(xt, wt, aggb, out);
}
